// TimeAwareMultiHeadAttention_29506425323613
// MI455X (gfx1250) — matrix operand order repaired and checked by reading, not compiled, not run on hardware
//
#include <hip/hip_runtime.h>
#include <hip/hip_bf16.h>

// B=2, S=256, H=256, NH=4, D=64, BS=512
#define Bb 2
#define Ss 256
#define Hh 256
#define NHh 4
#define Dd 64
#define BS 512
#define EPS 1e-12f

typedef __attribute__((ext_vector_type(16))) __bf16 v16bf;
typedef __attribute__((ext_vector_type(8)))  __bf16 v8bf;
typedef __attribute__((ext_vector_type(4)))  __bf16 v4bf;
typedef __attribute__((ext_vector_type(8)))  float  v8f;

__device__ __forceinline__ v8f wmma_bf16(v16bf a, v16bf b, v8f c) {
  return __builtin_amdgcn_wmma_f32_16x16x32_bf16(false, a, false, b, (short)0, c,
                                                 false, false);
}

// A fragment (16-bit A 16x32, ISA 7.12.2): two contiguous 16B runs per lane.
__device__ __forceinline__ v16bf ldA(const __bf16* row, int k0, int hi) {
  v8bf lo = *(const v8bf*)(row + k0 + hi * 8);
  v8bf hs = *(const v8bf*)(row + k0 + 16 + hi * 8);
  return __builtin_shufflevector(lo, hs, 0, 1, 2, 3, 4, 5, 6, 7,
                                 8, 9, 10, 11, 12, 13, 14, 15);
}
// B fragment from K-contiguous storage: lane = col N, two 8-K runs (same K order as ldA).
__device__ __forceinline__ v16bf ldB(const __bf16* colrow, int hi) {
  v8bf lo = *(const v8bf*)(colrow + hi * 8);
  v8bf hs = *(const v8bf*)(colrow + 16 + hi * 8);
  return __builtin_shufflevector(lo, hs, 0, 1, 2, 3, 4, 5, 6, 7,
                                 8, 9, 10, 11, 12, 13, 14, 15);
}

__device__ __forceinline__ unsigned lds_addr32(const void* p) {
  return (unsigned)(unsigned long long)p;   // low 32 bits of generic ptr = LDS byte addr
}
// CDNA5 async copy 16B global -> LDS (ASYNCcnt tracked)
__device__ __forceinline__ void async_cp16(unsigned lds, const void* g) {
  asm volatile("global_load_async_to_lds_b128 %0, %1, off"
               :: "v"(lds), "v"((unsigned long long)g) : "memory");
}
__device__ __forceinline__ void wait_async0() {
  asm volatile("s_wait_asynccnt 0" ::: "memory");
}

// ---------------------------------------------------------------------------
// Kernel W: transpose+convert all four weight matrices to bf16 WT[n][k].
// grid (8,8,4), 256 thr; coalesced read & write through a 32x33 LDS tile.
// ---------------------------------------------------------------------------
__global__ __launch_bounds__(256) void k_wt(
    const float* __restrict__ Wq, const float* __restrict__ Wk,
    const float* __restrict__ Wv, const float* __restrict__ Wd,
    __bf16* __restrict__ out) {
  __shared__ __bf16 tile[32][33];
  const int which = blockIdx.z;
  const float* W = (which == 0) ? Wq : (which == 1) ? Wk : (which == 2) ? Wv : Wd;
  __bf16* o = out + (size_t)which * Hh * Hh;
  const int k0 = blockIdx.x * 32, n0 = blockIdx.y * 32;
  const int c = threadIdx.x & 31, r8 = threadIdx.x >> 5;
#pragma unroll
  for (int i = 0; i < 4; ++i) {
    int r = r8 * 4 + i;
    tile[r][c] = (__bf16)W[(k0 + r) * Hh + n0 + c];
  }
  __syncthreads();
#pragma unroll
  for (int i = 0; i < 4; ++i) {
    int r = r8 * 4 + i;
    o[(n0 + r) * Hh + k0 + c] = tile[c][r];
  }
}

// ---------------------------------------------------------------------------
// Kernel A: fused QKV. B-slabs async double-buffered from bf16 WT.
// y=0: q (f32 [BS][H]); y=1: k+posK (bf16 [BS][H]); y=2: v+posV (bf16 [B][NH][D][S]).
// ---------------------------------------------------------------------------
__global__ __launch_bounds__(256) void k_qkv(
    const float* __restrict__ X, const __bf16* __restrict__ WT,
    const float* __restrict__ bq, const float* __restrict__ bk,
    const float* __restrict__ bv,
    const float* __restrict__ pK, const float* __restrict__ pV,
    float* __restrict__ outq, __bf16* __restrict__ outk, __bf16* __restrict__ outv) {
  __shared__ __align__(16) __bf16 x_lds[16][256];
  __shared__ __align__(16) __bf16 wT2[2][256][32];   // double-buffered B slab

  const int strip = blockIdx.x;
  const int which = blockIdx.y;
  const __bf16* WTm = WT + (size_t)which * Hh * Hh;
  const float* bias = (which == 0) ? bq : (which == 1) ? bk : bv;

  const int tid = threadIdx.x, lane = tid & 31, wv = tid >> 5;
  const int arow = lane & 15, hi = lane >> 4;

  // kick slab 0 fill (thread t owns WT row t: 32 halves = 4 x 16B)
  {
    const __bf16* g = WTm + (size_t)tid * Hh;
    unsigned l = lds_addr32(&wT2[0][tid][0]);
#pragma unroll
    for (int j = 0; j < 4; ++j) async_cp16(l + 16 * j, g + 8 * j);
  }
  // stage X strip (float4 loads, packed 8B bf16 stores)
#pragma unroll
  for (int i = 0; i < 4; ++i) {
    int v = tid + i * 256, m = v >> 6, n4 = (v & 63) * 4;
    float4 f = *(const float4*)&X[(strip * 16 + m) * Hh + n4];
    v4bf p; p[0] = (__bf16)f.x; p[1] = (__bf16)f.y; p[2] = (__bf16)f.z; p[3] = (__bf16)f.w;
    *(v4bf*)&x_lds[m][n4] = p;
  }

  v8f acc0 = {0, 0, 0, 0, 0, 0, 0, 0};
  v8f acc1 = acc0;
  int cur = 0;

  for (int s = 0; s < 8; ++s) {
    wait_async0();        // own wave's slab-s fill complete
    __syncthreads();      // all waves' fills complete; slab s-1 readers done
    if (s < 7) {          // fill next slab while computing on current
      const __bf16* g = WTm + (size_t)tid * Hh + (s + 1) * 32;
      unsigned l = lds_addr32(&wT2[cur ^ 1][tid][0]);
#pragma unroll
      for (int j = 0; j < 4; ++j) async_cp16(l + 16 * j, g + 8 * j);
    }
    v16bf a = ldA(&x_lds[arow][0], s * 32, hi);
#pragma unroll
    for (int t = 0; t < 2; ++t) {
      int n0 = (wv * 2 + t) * 16;
      v16bf bf = ldB(&wT2[cur][n0 + (lane & 15)][0], hi);
      if (t == 0) acc0 = wmma_bf16(a, bf, acc0);
      else        acc1 = wmma_bf16(a, bf, acc1);
    }
    cur ^= 1;
  }

#pragma unroll
  for (int t = 0; t < 2; ++t) {
    int col = (wv * 2 + t) * 16 + (lane & 15);
    float bval = bias[col];
    const v8f& ac = t ? acc1 : acc0;
#pragma unroll
    for (int r = 0; r < 8; ++r) {
      int grow = strip * 16 + hi * 8 + r;
      float v = ac[r] + bval;
      if (which == 0) {
        outq[grow * Hh + col] = v;
      } else if (which == 1) {
        v += pK[grow * Hh + col];
        outk[grow * Hh + col] = (__bf16)v;
      } else {
        v += pV[grow * Hh + col];
        int b = grow >> 8, key = grow & 255, h = col >> 6, d = col & 63;
        outv[(((b * NHh + h) * Dd) + d) * Ss + key] = (__bf16)v;  // transposed
      }
    }
  }
}

// ---------------------------------------------------------------------------
// Kernel B: attention for one (b, h, 16-query tile).
// ---------------------------------------------------------------------------
__global__ __launch_bounds__(256) void k_attn(
    const float* __restrict__ q_ws, const __bf16* __restrict__ kp_ws,
    const __bf16* __restrict__ vpT_ws, const float* __restrict__ mask,
    const float* __restrict__ tk, const float* __restrict__ tv,
    float* __restrict__ ctx_ws) {
  __shared__ __align__(16) __bf16 kbuf[16384];  // kp [256][64] -> vpT [64][256] (32 KB)
  __shared__ __align__(16) float  S[16][256];   // scores -> probs f32 (16 KB)
  __shared__ __align__(16) __bf16 Pb[16][256];  // probs bf16          ( 8 KB)
  __shared__ __align__(16) float  qc[16][64];   // q f32, reused as ctx( 4 KB)
  __shared__ __align__(16) __bf16 q16[16][64];  // q bf16 for WMMA     ( 2 KB)

  const int qt = blockIdx.x, h = blockIdx.y, b = blockIdx.z;
  const int q0 = qt * 16;
  const int tid = threadIdx.x, lane = tid & 31, wv = tid >> 5;
  const int arow = lane & 15, hi = lane >> 4;
  const int rowbase = b * Ss;
  const int hcol = h * Dd;

  // async fill of kp tile: thread t copies key row t (64 bf16 = 8 x 16B)
  {
    const __bf16* g = kp_ws + (size_t)(rowbase + tid) * Hh + hcol;
    unsigned l = lds_addr32(&kbuf[tid * 64]);
#pragma unroll
    for (int j = 0; j < 8; ++j) async_cp16(l + 16 * j, g + 8 * j);
  }
  // stage q tile (f32 + bf16 copies): one float4 per thread
  {
    int m = tid >> 4, d4 = (tid & 15) * 4;
    float4 f = *(const float4*)&q_ws[(rowbase + q0 + m) * Hh + hcol + d4];
    *(float4*)&qc[m][d4] = f;
    v4bf p; p[0] = (__bf16)f.x; p[1] = (__bf16)f.y; p[2] = (__bf16)f.z; p[3] = (__bf16)f.w;
    *(v4bf*)&q16[m][d4] = p;
  }
  wait_async0();
  __syncthreads();

  // ---- phase 1a: S = q * kp^T via WMMA (8 waves x 2 key tiles) ----
#pragma unroll
  for (int t = 0; t < 2; ++t) {
    int n0 = (wv * 2 + t) * 16;
    v8f acc = {0, 0, 0, 0, 0, 0, 0, 0};
#pragma unroll
    for (int k0 = 0; k0 < 64; k0 += 32) {
      v16bf a  = ldA(&q16[arow][0], k0, hi);
      v16bf bf = ldB(&kbuf[(n0 + (lane & 15)) * 64 + k0], hi);
      acc = wmma_bf16(a, bf, acc);
    }
#pragma unroll
    for (int r = 0; r < 8; ++r) S[r + hi * 8][n0 + (lane & 15)] = acc[r];
  }
  __syncthreads();

  // ---- phase 1b: S[i][k] += dot(tk[b,q0+i,k,h,:], q), thread owns key k ----
  {
    const int k = tid;
#pragma unroll 2
    for (int i = 0; i < 16; ++i) {
      const float4* tp = (const float4*)(tk +
          (((size_t)(b * Ss + q0 + i) * Ss + k) * Hh + hcol));
      __builtin_prefetch(tp + (Ss * Hh) / 4, 0, 1);   // row for i+1
      float dot = 0.f;
#pragma unroll
      for (int j4 = 0; j4 < 16; ++j4) {
        float4 tq = tp[j4];
        dot += tq.x * qc[i][j4 * 4 + 0] + tq.y * qc[i][j4 * 4 + 1] +
               tq.z * qc[i][j4 * 4 + 2] + tq.w * qc[i][j4 * 4 + 3];
      }
      S[i][k] += dot;
    }
  }
  __syncthreads();

  // ---- phase 2: scale + mask + softmax; lane owns 8 contiguous cols ----
#pragma unroll
  for (int rr = 0; rr < 2; ++rr) {
    int row = wv * 2 + rr;
    const float* mrow = mask + ((size_t)(b * Ss + q0 + row)) * Ss;
    float x[8];
    {
      float4 xa = *(const float4*)&S[row][lane * 8];
      float4 xb = *(const float4*)&S[row][lane * 8 + 4];
      float4 ma = *(const float4*)&mrow[lane * 8];
      float4 mb = *(const float4*)&mrow[lane * 8 + 4];
      x[0] = xa.x * 0.125f + ma.x; x[1] = xa.y * 0.125f + ma.y;
      x[2] = xa.z * 0.125f + ma.z; x[3] = xa.w * 0.125f + ma.w;
      x[4] = xb.x * 0.125f + mb.x; x[5] = xb.y * 0.125f + mb.y;
      x[6] = xb.z * 0.125f + mb.z; x[7] = xb.w * 0.125f + mb.w;
    }
    float mx = x[0];
#pragma unroll
    for (int j = 1; j < 8; ++j) mx = fmaxf(mx, x[j]);
    for (int off = 16; off; off >>= 1) mx = fmaxf(mx, __shfl_xor(mx, off, 32));
    float sum = 0.f;
#pragma unroll
    for (int j = 0; j < 8; ++j) { x[j] = __expf(x[j] - mx); sum += x[j]; }
    for (int off = 16; off; off >>= 1) sum += __shfl_xor(sum, off, 32);
    float inv = 1.0f / sum;
    v8bf pv;
#pragma unroll
    for (int j = 0; j < 8; ++j) { x[j] *= inv; pv[j] = (__bf16)x[j]; }
    *(float4*)&S[row][lane * 8]     = make_float4(x[0], x[1], x[2], x[3]);
    *(float4*)&S[row][lane * 8 + 4] = make_float4(x[4], x[5], x[6], x[7]);
    *(v8bf*)&Pb[row][lane * 8] = pv;
  }
  __syncthreads();

  // ---- phase 3a: kick async vpT fill, overlap with P.tv stream ----
  {
    const __bf16* g = vpT_ws + ((size_t)(b * NHh + h) * Dd + (tid >> 2)) * Ss + (tid & 3) * 64;
    unsigned l = lds_addr32(&kbuf[(tid >> 2) * 256 + (tid & 3) * 64]);
#pragma unroll
    for (int j = 0; j < 8; ++j) async_cp16(l + 16 * j, g + 8 * j);
  }
  {
    int row = wv * 2 + hi;          // half-wave owns one row
    int d4 = (lane & 15) * 4;
    size_t base = ((size_t)(b * Ss + q0 + row)) * Ss * Hh + hcol + d4;
    float4 acc = make_float4(0.f, 0.f, 0.f, 0.f);
    for (int k = 0; k < 256; ++k) {
      __builtin_prefetch(tv + base + (size_t)(k + 8) * Hh, 0, 1);
      float p = S[row][k];
      float4 t4 = *(const float4*)(tv + base + (size_t)k * Hh);
      acc.x = fmaf(p, t4.x, acc.x); acc.y = fmaf(p, t4.y, acc.y);
      acc.z = fmaf(p, t4.z, acc.z); acc.w = fmaf(p, t4.w, acc.w);
    }
    *(float4*)&qc[row][d4] = acc;   // qc now = ctx (tv part)
  }
  wait_async0();
  __syncthreads();

  // ---- phase 3b: ctx += P * vp via WMMA (4 d-tiles x 2 key-halves) ----
  {
    int n0 = (wv & 3) * 16;
    int kh = (wv >> 2) * 128;
    v8f acc = {0, 0, 0, 0, 0, 0, 0, 0};
#pragma unroll
    for (int ks = 0; ks < 4; ++ks) {
      int kslice = kh + ks * 32;
      v16bf a  = ldA(&Pb[arow][0], kslice, hi);
      v16bf bf = ldB(&kbuf[(n0 + (lane & 15)) * 256 + kslice], hi);  // vpT[d][key]
      acc = wmma_bf16(a, bf, acc);
    }
#pragma unroll
    for (int r = 0; r < 8; ++r)
      atomicAdd(&qc[r + hi * 8][n0 + (lane & 15)], acc[r]);  // ds_add_f32 merge
  }
  __syncthreads();

  {
    int m = tid >> 4, d4 = (tid & 15) * 4;
    *(float4*)&ctx_ws[(rowbase + q0 + m) * Hh + hcol + d4] = *(float4*)&qc[m][d4];
  }
}

// ---------------------------------------------------------------------------
// Kernel C: h = ctx @ Wd + bd + input ; LayerNorm(h) -> out (fused per strip).
// ---------------------------------------------------------------------------
__global__ __launch_bounds__(256) void k_out(
    const float* __restrict__ ctx_ws, const __bf16* __restrict__ WdT,
    const float* __restrict__ bd, const float* __restrict__ X,
    const float* __restrict__ gamma, const float* __restrict__ beta,
    float* __restrict__ out) {
  __shared__ __align__(16) __bf16 a_lds[16][256];
  __shared__ __align__(16) __bf16 wT2[2][256][32];
  __shared__ __align__(16) float  h_lds[16][256];

  const int strip = blockIdx.x;
  const int tid = threadIdx.x, lane = tid & 31, wv = tid >> 5;
  const int arow = lane & 15, hi = lane >> 4;

  {
    const __bf16* g = WdT + (size_t)tid * Hh;
    unsigned l = lds_addr32(&wT2[0][tid][0]);
#pragma unroll
    for (int j = 0; j < 4; ++j) async_cp16(l + 16 * j, g + 8 * j);
  }
#pragma unroll
  for (int i = 0; i < 4; ++i) {
    int v = tid + i * 256, m = v >> 6, n4 = (v & 63) * 4;
    float4 f = *(const float4*)&ctx_ws[(strip * 16 + m) * Hh + n4];
    v4bf p; p[0] = (__bf16)f.x; p[1] = (__bf16)f.y; p[2] = (__bf16)f.z; p[3] = (__bf16)f.w;
    *(v4bf*)&a_lds[m][n4] = p;
  }

  v8f acc0 = {0, 0, 0, 0, 0, 0, 0, 0};
  v8f acc1 = acc0;
  int cur = 0;

  for (int s = 0; s < 8; ++s) {
    wait_async0();
    __syncthreads();
    if (s < 7) {
      const __bf16* g = WdT + (size_t)tid * Hh + (s + 1) * 32;
      unsigned l = lds_addr32(&wT2[cur ^ 1][tid][0]);
#pragma unroll
      for (int j = 0; j < 4; ++j) async_cp16(l + 16 * j, g + 8 * j);
    }
    v16bf a = ldA(&a_lds[arow][0], s * 32, hi);
#pragma unroll
    for (int t = 0; t < 2; ++t) {
      int n0 = (wv * 2 + t) * 16;
      v16bf bf = ldB(&wT2[cur][n0 + (lane & 15)][0], hi);
      if (t == 0) acc0 = wmma_bf16(a, bf, acc0);
      else        acc1 = wmma_bf16(a, bf, acc1);
    }
    cur ^= 1;
  }

#pragma unroll
  for (int t = 0; t < 2; ++t) {
    int col = (wv * 2 + t) * 16 + (lane & 15);
    float bval = bd[col];
    const v8f& ac = t ? acc1 : acc0;
#pragma unroll
    for (int r = 0; r < 8; ++r) {
      int lrow = hi * 8 + r;
      h_lds[lrow][col] = ac[r] + bval + X[(strip * 16 + lrow) * Hh + col];
    }
  }
  __syncthreads();

  // LayerNorm per row; lane owns 8 contiguous cols
#pragma unroll
  for (int rr = 0; rr < 2; ++rr) {
    int row = wv * 2 + rr;
    int grow = strip * 16 + row;
    float4 xa = *(const float4*)&h_lds[row][lane * 8];
    float4 xb = *(const float4*)&h_lds[row][lane * 8 + 4];
    float x[8] = {xa.x, xa.y, xa.z, xa.w, xb.x, xb.y, xb.z, xb.w};
    float sum = 0.f;
#pragma unroll
    for (int j = 0; j < 8; ++j) sum += x[j];
    for (int off = 16; off; off >>= 1) sum += __shfl_xor(sum, off, 32);
    float mu = sum * (1.0f / 256.0f);
    float vs = 0.f;
#pragma unroll
    for (int j = 0; j < 8; ++j) { float d = x[j] - mu; vs += d * d; }
    for (int off = 16; off; off >>= 1) vs += __shfl_xor(vs, off, 32);
    float rstd = rsqrtf(vs * (1.0f / 256.0f) + EPS);
    float4 ga = *(const float4*)&gamma[lane * 8];
    float4 gb = *(const float4*)&gamma[lane * 8 + 4];
    float4 ba = *(const float4*)&beta[lane * 8];
    float4 bb2 = *(const float4*)&beta[lane * 8 + 4];
    float4 oa = make_float4((x[0] - mu) * rstd * ga.x + ba.x,
                            (x[1] - mu) * rstd * ga.y + ba.y,
                            (x[2] - mu) * rstd * ga.z + ba.z,
                            (x[3] - mu) * rstd * ga.w + ba.w);
    float4 ob = make_float4((x[4] - mu) * rstd * gb.x + bb2.x,
                            (x[5] - mu) * rstd * gb.y + bb2.y,
                            (x[6] - mu) * rstd * gb.z + bb2.z,
                            (x[7] - mu) * rstd * gb.w + bb2.w);
    *(float4*)&out[grow * Hh + lane * 8]     = oa;
    *(float4*)&out[grow * Hh + lane * 8 + 4] = ob;
  }
}

// ---------------------------------------------------------------------------
extern "C" void kernel_launch(void* const* d_in, const int* in_sizes, int n_in,
                              void* d_out, int out_size, void* d_ws, size_t ws_size,
                              hipStream_t stream) {
  (void)in_sizes; (void)n_in; (void)out_size; (void)ws_size;
  const float* X    = (const float*)d_in[0];
  const float* mask = (const float*)d_in[1];
  const float* pK   = (const float*)d_in[2];
  const float* pV   = (const float*)d_in[3];
  const float* tk   = (const float*)d_in[4];
  const float* tv   = (const float*)d_in[5];
  const float* Wq   = (const float*)d_in[6];
  const float* bq   = (const float*)d_in[7];
  const float* Wk   = (const float*)d_in[8];
  const float* bk   = (const float*)d_in[9];
  const float* Wv   = (const float*)d_in[10];
  const float* bv   = (const float*)d_in[11];
  const float* Wd   = (const float*)d_in[12];
  const float* bd   = (const float*)d_in[13];
  const float* gm   = (const float*)d_in[14];
  const float* bt   = (const float*)d_in[15];

  float*  ws   = (float*)d_ws;
  float*  qb   = ws;                                     // f32 [BS][H]
  __bf16* kpb  = (__bf16*)(ws + BS * Hh);                // bf16 [BS][H]
  __bf16* vpb  = (__bf16*)(ws + BS * Hh + BS * Hh / 2);  // bf16 [B][NH][D][S]
  float*  ctxb = ws + 2 * BS * Hh;                       // f32 [BS][H]
  __bf16* wtb  = (__bf16*)(ws + 3 * BS * Hh);            // bf16 [4][H][H] (N-major)

  k_wt<<<dim3(8, 8, 4), 256, 0, stream>>>(Wq, Wk, Wv, Wd, wtb);
  k_qkv<<<dim3(BS / 16, 3, 1), 256, 0, stream>>>(X, wtb, bq, bk, bv,
                                                 pK, pV, qb, kpb, vpb);
  k_attn<<<dim3(Ss / 16, NHh, Bb), 256, 0, stream>>>(qb, kpb, vpb, mask, tk, tv,
                                                     ctxb);
  k_out<<<dim3(BS / 16, 1, 1), 256, 0, stream>>>(ctxb, wtb + 3 * Hh * Hh, bd, X,
                                                 gm, bt, (float*)d_out);
}
